// TransformerBlock_77472620085643
// MI455X (gfx1250) — matrix operand order repaired and checked by reading, not compiled, not run on hardware
//
#include <hip/hip_runtime.h>
#include <hip/hip_bf16.h>
#include <stdint.h>

#define B_   2
#define S_   1024
#define H_   1024
#define NH_  16
#define KVH_ 4
#define HD_  64
#define E_   8
#define I_   2048
#define T_   (B_*S_)
#define WIN_ 512

typedef __attribute__((ext_vector_type(16))) __bf16 bf16x16;
typedef __attribute__((ext_vector_type(8)))  float  f32x8;

#define WMMA_BF16(a,b,c) __builtin_amdgcn_wmma_f32_16x16x32_bf16(false,(a),false,(b),(short)0,(c),false,false)

__device__ __forceinline__ uint16_t f2bf(float f) {
  uint32_t u = __float_as_uint(f);
  uint32_t r = u + 0x7FFFu + ((u >> 16) & 1u);
  return (uint16_t)(r >> 16);
}
__device__ __forceinline__ float bf2f(uint16_t h) {
  return __uint_as_float(((uint32_t)h) << 16);
}

union FragU { uint4 q[2]; bf16x16 v; };

// A-matrix 16x32 bf16 fragment (row-major source). elems 0..7 -> K = kk+half*8+e,
// elems 8..15 -> K = kk+16+half*8+e  (per CDNA5 ISA 16-bit A layout).
__device__ __forceinline__ bf16x16 load_frag_A(const uint16_t* row, int kk, int half) {
  FragU f;
  f.q[0] = *(const uint4*)(row + kk + half * 8);
  f.q[1] = *(const uint4*)(row + kk + 16 + half * 8);
  return f.v;
}
// B-matrix 32x16 bf16 fragment from Bt[N,K] (K contiguous). elems 0..7 -> K = kk+half*8+e,
// elems 8..15 -> K = kk+16+half*8+e.
__device__ __forceinline__ bf16x16 load_frag_B(const uint16_t* row, int kk, int half) {
  FragU f;
  f.q[0] = *(const uint4*)(row + kk + half * 8);
  f.q[1] = *(const uint4*)(row + kk + 16 + half * 8);
  return f.v;
}

// ---------------- RMSNorm (fp32 in -> bf16 out) ----------------
__global__ __launch_bounds__(256) void rmsnorm_bf16_k(const float* __restrict__ X,
                                                      const float* __restrict__ W,
                                                      uint16_t* __restrict__ Y) {
  int row = blockIdx.x, tid = threadIdx.x;
  const float* xr = X + (size_t)row * H_;
  float ss = 0.f;
  for (int i = tid; i < H_; i += 256) { float v = xr[i]; ss += v * v; }
  __shared__ float red[256];
  red[tid] = ss; __syncthreads();
  for (int s = 128; s > 0; s >>= 1) { if (tid < s) red[tid] += red[tid + s]; __syncthreads(); }
  float inv = rsqrtf(red[0] / (float)H_ + 1e-5f);
  for (int i = tid; i < H_; i += 256) Y[(size_t)row * H_ + i] = f2bf(xr[i] * inv * W[i]);
}

// ---------------- fp32 [K,N] -> bf16 transposed [N,K] ----------------
__global__ __launch_bounds__(256) void convT_bf16_k(const float* __restrict__ src,
                                                    uint16_t* __restrict__ dst, int K, int N) {
  int idx = blockIdx.x * 256 + threadIdx.x;
  if (idx >= K * N) return;
  int k = idx / N, n = idx - k * N;
  dst[(size_t)n * K + k] = f2bf(src[idx]);
}

// ---------------- blocked bf16 WMMA GEMM: wave = 32x64 tile of C = A @ Bt^T (+res) ----------
template <int ADD_RES>
__global__ __launch_bounds__(256) void gemm_bf16_f32_k(const uint16_t* __restrict__ A,
                                                       const uint16_t* __restrict__ Bt,
                                                       const float* __restrict__ Res,
                                                       float* __restrict__ C,
                                                       int M, int N, int K) {
  int lane = threadIdx.x, half = lane >> 4, l = lane & 15;
  int sn = blockIdx.x;                       // N/64 supertiles
  int sm = blockIdx.y * 8 + threadIdx.y;     // M/32 supertiles
  if (sm * 32 >= M) return;
  const uint16_t* ar0 = A + (size_t)(sm * 32 + l) * K;
  const uint16_t* ar1 = ar0 + (size_t)16 * K;
  const uint16_t* br[4];
#pragma unroll
  for (int j = 0; j < 4; ++j) br[j] = Bt + (size_t)(sn * 64 + j * 16 + l) * K;
  f32x8 acc[2][4] = {};
#pragma unroll 2
  for (int kk = 0; kk < K; kk += 32) {
    bf16x16 a0 = load_frag_A(ar0, kk, half);
    bf16x16 a1 = load_frag_A(ar1, kk, half);
    __builtin_prefetch((const void*)(ar0 + kk + 64), 0, 1);
    __builtin_prefetch((const void*)(ar1 + kk + 64), 0, 1);
#pragma unroll
    for (int j = 0; j < 4; ++j) {
      bf16x16 b = load_frag_B(br[j], kk, half);
      __builtin_prefetch((const void*)(br[j] + kk + 64), 0, 1);
      acc[0][j] = WMMA_BF16(a0, b, acc[0][j]);
      acc[1][j] = WMMA_BF16(a1, b, acc[1][j]);
    }
  }
#pragma unroll
  for (int mt = 0; mt < 2; ++mt)
#pragma unroll
    for (int j = 0; j < 4; ++j)
#pragma unroll
      for (int r = 0; r < 8; ++r) {
        int row = sm * 32 + mt * 16 + r + 8 * half;
        int col = sn * 64 + j * 16 + l;
        float v = acc[mt][j][r];
        if (ADD_RES) v += Res[(size_t)row * N + col];
        C[(size_t)row * N + col] = v;
      }
}

// ---------------- RoPE Q (folds 1/sqrt(HD)), fp32 -> bf16 [B,NH,S,HD] ----------------
__global__ __launch_bounds__(256) void rope_q_k(const float* __restrict__ qf,
                                                uint16_t* __restrict__ qb) {
  int idx = blockIdx.x * 256 + threadIdx.x;
  int i = idx & 31, nh = (idx >> 5) & 15, s = (idx >> 9) & 1023, b = idx >> 19;
  float inv = __powf(10000.f, -2.f * (float)i / 64.f);
  float sn, cs; __sincosf((float)s * inv, &sn, &cs);
  const float* base = qf + ((size_t)(b * S_ + s)) * (NH_ * HD_) + nh * HD_;
  float x1 = base[i], x2 = base[i + 32];
  uint16_t* ob = qb + ((size_t)(b * NH_ + nh) * S_ + s) * HD_;
  const float scale = 0.125f; // 1/sqrt(64)
  ob[i]      = f2bf((x1 * cs - x2 * sn) * scale);
  ob[i + 32] = f2bf((x2 * cs + x1 * sn) * scale);
}

// ---------------- RoPE K, fp32 -> bf16 [B,KVH,S,HD] ----------------
__global__ __launch_bounds__(256) void rope_k_k(const float* __restrict__ kf,
                                                uint16_t* __restrict__ kb) {
  int idx = blockIdx.x * 256 + threadIdx.x;
  int i = idx & 31, kvh = (idx >> 5) & 3, s = (idx >> 7) & 1023, b = idx >> 17;
  float inv = __powf(10000.f, -2.f * (float)i / 64.f);
  float sn, cs; __sincosf((float)s * inv, &sn, &cs);
  const float* base = kf + ((size_t)(b * S_ + s)) * (KVH_ * HD_) + kvh * HD_;
  float x1 = base[i], x2 = base[i + 32];
  uint16_t* ob = kb + ((size_t)(b * KVH_ + kvh) * S_ + s) * HD_;
  ob[i]      = f2bf(x1 * cs - x2 * sn);
  ob[i + 32] = f2bf(x2 * cs + x1 * sn);
}

// ---------------- V transpose: fp32 [B,S,KVH*HD] -> bf16 [B,KVH,HD,S] ----------------
__global__ __launch_bounds__(256) void vtrans_k(const float* __restrict__ vf,
                                                uint16_t* __restrict__ vt) {
  int idx = blockIdx.x * 256 + threadIdx.x;
  int d = idx & 63, kvh = (idx >> 6) & 3, s = (idx >> 8) & 1023, b = idx >> 18;
  float v = vf[((size_t)(b * S_ + s)) * (KVH_ * HD_) + kvh * HD_ + d];
  vt[((size_t)(b * KVH_ + kvh) * HD_ + d) * S_ + s] = f2bf(v);
}

// ---------------- flash attention, one wave = one (b,h,16-query tile) ----------------
__global__ __launch_bounds__(256) void attention_k(const uint16_t* __restrict__ qb,
                                                   const uint16_t* __restrict__ kb,
                                                   const uint16_t* __restrict__ vt,
                                                   uint16_t* __restrict__ attn) {
  __shared__ __align__(16) uint16_t pbuf[8][16][32];
  int lane = threadIdx.x, half = lane >> 4, l = lane & 15, w = threadIdx.y;
  int wid = blockIdx.x * 8 + w;
  int qt = wid & 63, h = (wid >> 6) & 15, b = wid >> 10;
  int kvh = h >> 2;

  const uint16_t* qbase = qb + ((size_t)(b * NH_ + h) * S_) * HD_;
  const uint16_t* kbase = kb + ((size_t)(b * KVH_ + kvh) * S_) * HD_;
  const uint16_t* vbase = vt + ((size_t)(b * KVH_ + kvh) * HD_) * S_;

  const uint16_t* qrow = qbase + (size_t)(qt * 16 + l) * HD_;
  bf16x16 qa0 = load_frag_A(qrow, 0, half);
  bf16x16 qa1 = load_frag_A(qrow, 32, half);

  f32x8 o[4] = {};
  float rmax[8], rsum[8];
#pragma unroll
  for (int r = 0; r < 8; ++r) { rmax[r] = -1e30f; rsum[r] = 0.f; }

  int startk = (qt >= 32) ? (qt - 32) * 16 : 0;
  int endk = qt * 16;
  for (int kk = startk; kk <= endk; kk += 32) {
    f32x8 s0 = {}, s1 = {};
    { // keys kk..kk+15
      const uint16_t* krow = kbase + (size_t)(kk + l) * HD_;
      s0 = WMMA_BF16(qa0, load_frag_B(krow, 0, half), s0);
      s0 = WMMA_BF16(qa1, load_frag_B(krow, 32, half), s0);
    }
    { // keys kk+16..kk+31 (clamped load, masked below)
      int key = kk + 16 + l; if (key > S_ - 1) key = S_ - 1;
      const uint16_t* krow = kbase + (size_t)key * HD_;
      s1 = WMMA_BF16(qa0, load_frag_B(krow, 0, half), s1);
      s1 = WMMA_BF16(qa1, load_frag_B(krow, 32, half), s1);
    }
    // mask + online softmax
#pragma unroll
    for (int r = 0; r < 8; ++r) {
      int qi = qt * 16 + r + 8 * half;
      int k0 = kk + l, k1 = kk + 16 + l;
      if (k0 > qi || qi - k0 > WIN_) s0[r] = -1e30f;
      if (k1 > qi || qi - k1 > WIN_) s1[r] = -1e30f;
    }
#pragma unroll
    for (int r = 0; r < 8; ++r) {
      float mloc = fmaxf(s0[r], s1[r]);
      for (int off = 1; off < 16; off <<= 1) mloc = fmaxf(mloc, __shfl_xor(mloc, off, 32));
      float nm = fmaxf(rmax[r], mloc);
      float sc = __expf(rmax[r] - nm);
      float p0 = __expf(s0[r] - nm);
      float p1 = __expf(s1[r] - nm);
      float ps = p0 + p1;
      for (int off = 1; off < 16; off <<= 1) ps += __shfl_xor(ps, off, 32);
      rsum[r] = rsum[r] * sc + ps;
      rmax[r] = nm;
#pragma unroll
      for (int c = 0; c < 4; ++c) o[c][r] *= sc;
      int row = r + 8 * half;
      pbuf[w][row][l]      = f2bf(p0);
      pbuf[w][row][16 + l] = f2bf(p1);
    }
    // P (C-layout in LDS) -> A-layout fragment
    const uint16_t* prow = &pbuf[w][l][0];
    FragU pf;
    pf.q[0] = *(const uint4*)(prow + half * 8);
    pf.q[1] = *(const uint4*)(prow + 16 + half * 8);
    int kb0 = kk + half * 8, kb1 = kk + 16 + half * 8; if (kb1 > S_ - 8) kb1 = S_ - 8;
#pragma unroll
    for (int c = 0; c < 4; ++c) {
      const uint16_t* vrow = vbase + (size_t)(c * 16 + l) * S_;
      FragU vfr;
      vfr.q[0] = *(const uint4*)(vrow + kb0);
      vfr.q[1] = *(const uint4*)(vrow + kb1);
      o[c] = WMMA_BF16(pf.v, vfr.v, o[c]);
    }
  }
#pragma unroll
  for (int c = 0; c < 4; ++c)
#pragma unroll
    for (int r = 0; r < 8; ++r) {
      int qi = qt * 16 + r + 8 * half;
      attn[((size_t)(b * S_ + qi)) * (NH_ * HD_) + h * HD_ + c * 16 + l] = f2bf(o[c][r] / rsum[r]);
    }
}

// ---------------- MoE gate: softmax + top-2, compact per-expert lists ----------------
__global__ void zero_counts_k(int* cnt) { if (blockIdx.x == 0 && threadIdx.x < E_) cnt[threadIdx.x] = 0; }

__global__ __launch_bounds__(256) void gate_topk_k(const uint16_t* __restrict__ xn2,
                                                   const float* __restrict__ gw,
                                                   int* __restrict__ cnt,
                                                   int* __restrict__ toklist,
                                                   int* __restrict__ slotlist,
                                                   float* __restrict__ wlist) {
  int tok = blockIdx.x * 8 + threadIdx.y;
  int lane = threadIdx.x;
  if (tok >= T_) return;
  const uint16_t* xr = xn2 + (size_t)tok * H_;
  float acc[E_] = {};
  for (int hh = lane; hh < H_; hh += 32) {
    float xv = bf2f(xr[hh]);
    const float* g = gw + (size_t)hh * E_;
#pragma unroll
    for (int e = 0; e < E_; ++e) acc[e] += xv * g[e];
  }
#pragma unroll
  for (int e = 0; e < E_; ++e)
    for (int off = 16; off > 0; off >>= 1) acc[e] += __shfl_xor(acc[e], off, 32);
  if (lane == 0) {
    float mx = acc[0];
#pragma unroll
    for (int e = 1; e < E_; ++e) mx = fmaxf(mx, acc[e]);
    float p[E_];
#pragma unroll
    for (int e = 0; e < E_; ++e) p[e] = __expf(acc[e] - mx);
    int e0 = 0; float b0 = p[0];
#pragma unroll
    for (int e = 1; e < E_; ++e) if (p[e] > b0) { b0 = p[e]; e0 = e; }
    int e1 = (e0 == 0) ? 1 : 0; float b1 = p[e1];
#pragma unroll
    for (int e = 0; e < E_; ++e) if (e != e0 && p[e] > b1) { b1 = p[e]; e1 = e; }
    float w0 = b0 / (b0 + b1), w1 = b1 / (b0 + b1);
    int pos0 = atomicAdd(&cnt[e0], 1);
    toklist[e0 * T_ + pos0] = tok; slotlist[e0 * T_ + pos0] = 0; wlist[e0 * T_ + pos0] = w0;
    int pos1 = atomicAdd(&cnt[e1], 1);
    toklist[e1 * T_ + pos1] = tok; slotlist[e1 * T_ + pos1] = 1; wlist[e1 * T_ + pos1] = w1;
  }
}

// ------- gathered gate+up GEMM, wave = 32 rows x 32 cols x 2 mats, SiLU-mul -> bf16 -------
__global__ __launch_bounds__(256) void gemm_gateup_k(const uint16_t* __restrict__ Xn,
                                                     const int* __restrict__ rowidx,
                                                     const int* __restrict__ cntPtr,
                                                     const uint16_t* __restrict__ B1t,
                                                     const uint16_t* __restrict__ B2t,
                                                     uint16_t* __restrict__ Hg,
                                                     int N, int K) {
  int cnt = *cntPtr;
  int lane = threadIdx.x, half = lane >> 4, l = lane & 15;
  int sn = blockIdx.x;                    // N/32
  int sm = blockIdx.y * 8 + threadIdx.y;  // M/32
  if (sm * 32 >= cnt) return;
  int ar0i = sm * 32 + l;      if (ar0i > cnt - 1) ar0i = cnt - 1;
  int ar1i = sm * 32 + 16 + l; if (ar1i > cnt - 1) ar1i = cnt - 1;
  const uint16_t* ar0 = Xn + (size_t)rowidx[ar0i] * K;
  const uint16_t* ar1 = Xn + (size_t)rowidx[ar1i] * K;
  const uint16_t* bg[2]; const uint16_t* bu[2];
#pragma unroll
  for (int j = 0; j < 2; ++j) {
    bg[j] = B1t + (size_t)(sn * 32 + j * 16 + l) * K;
    bu[j] = B2t + (size_t)(sn * 32 + j * 16 + l) * K;
  }
  f32x8 cg[2][2] = {}, cu[2][2] = {};
#pragma unroll 2
  for (int kk = 0; kk < K; kk += 32) {
    bf16x16 a0 = load_frag_A(ar0, kk, half);
    bf16x16 a1 = load_frag_A(ar1, kk, half);
#pragma unroll
    for (int j = 0; j < 2; ++j) {
      bf16x16 g = load_frag_B(bg[j], kk, half);
      bf16x16 u = load_frag_B(bu[j], kk, half);
      __builtin_prefetch((const void*)(bg[j] + kk + 64), 0, 1);
      __builtin_prefetch((const void*)(bu[j] + kk + 64), 0, 1);
      cg[0][j] = WMMA_BF16(a0, g, cg[0][j]);
      cg[1][j] = WMMA_BF16(a1, g, cg[1][j]);
      cu[0][j] = WMMA_BF16(a0, u, cu[0][j]);
      cu[1][j] = WMMA_BF16(a1, u, cu[1][j]);
    }
  }
#pragma unroll
  for (int mt = 0; mt < 2; ++mt)
#pragma unroll
    for (int j = 0; j < 2; ++j)
#pragma unroll
      for (int r = 0; r < 8; ++r) {
        int row = sm * 32 + mt * 16 + r + 8 * half;
        if (row < cnt) {
          float g = cg[mt][j][r], u = cu[mt][j][r];
          float s = g / (1.f + __expf(-g));
          Hg[(size_t)row * N + sn * 32 + j * 16 + l] = f2bf(s * u);
        }
      }
}

// ------- down GEMM, wave = 32x64 tile, weighted scatter into per-slot buffers -------
__global__ __launch_bounds__(256) void gemm_down_k(const uint16_t* __restrict__ Hg,
                                                   const int* __restrict__ cntPtr,
                                                   const int* __restrict__ toklist,
                                                   const int* __restrict__ slotlist,
                                                   const float* __restrict__ wlist,
                                                   const uint16_t* __restrict__ Bt,
                                                   float* __restrict__ outslot,
                                                   int N, int K) {
  int cnt = *cntPtr;
  int lane = threadIdx.x, half = lane >> 4, l = lane & 15;
  int sn = blockIdx.x;                    // N/64
  int sm = blockIdx.y * 8 + threadIdx.y;  // M/32
  if (sm * 32 >= cnt) return;
  const uint16_t* ar0 = Hg + (size_t)(sm * 32 + l) * K;
  const uint16_t* ar1 = ar0 + (size_t)16 * K;
  const uint16_t* br[4];
#pragma unroll
  for (int j = 0; j < 4; ++j) br[j] = Bt + (size_t)(sn * 64 + j * 16 + l) * K;
  f32x8 acc[2][4] = {};
#pragma unroll 2
  for (int kk = 0; kk < K; kk += 32) {
    bf16x16 a0 = load_frag_A(ar0, kk, half);
    bf16x16 a1 = load_frag_A(ar1, kk, half);
    __builtin_prefetch((const void*)(ar0 + kk + 64), 0, 1);
    __builtin_prefetch((const void*)(ar1 + kk + 64), 0, 1);
#pragma unroll
    for (int j = 0; j < 4; ++j) {
      bf16x16 b = load_frag_B(br[j], kk, half);
      __builtin_prefetch((const void*)(br[j] + kk + 64), 0, 1);
      acc[0][j] = WMMA_BF16(a0, b, acc[0][j]);
      acc[1][j] = WMMA_BF16(a1, b, acc[1][j]);
    }
  }
#pragma unroll
  for (int mt = 0; mt < 2; ++mt)
#pragma unroll
    for (int j = 0; j < 4; ++j)
#pragma unroll
      for (int r = 0; r < 8; ++r) {
        int row = sm * 32 + mt * 16 + r + 8 * half;
        if (row < cnt) {
          int tok = toklist[row], sl = slotlist[row];
          outslot[((size_t)sl * T_ + tok) * N + sn * 64 + j * 16 + l] = acc[mt][j][r] * wlist[row];
        }
      }
}

// ---------------- final: out = x1 + slot0 + slot1 ----------------
__global__ __launch_bounds__(256) void final_combine_k(const float* __restrict__ x1,
                                                       const float* __restrict__ outslot,
                                                       float* __restrict__ out) {
  int i = blockIdx.x * 256 + threadIdx.x;
  if (i >= T_ * H_) return;
  out[i] = x1[i] + outslot[i] + outslot[(size_t)T_ * H_ + i];
}

extern "C" void kernel_launch(void* const* d_in, const int* in_sizes, int n_in,
                              void* d_out, int out_size, void* d_ws, size_t ws_size,
                              hipStream_t stream) {
  (void)in_sizes; (void)n_in; (void)out_size; (void)ws_size;
  const float* x      = (const float*)d_in[0];
  const float* anw    = (const float*)d_in[1];
  const float* mnw    = (const float*)d_in[2];
  const float* wq     = (const float*)d_in[3];
  const float* wk     = (const float*)d_in[4];
  const float* wv     = (const float*)d_in[5];
  const float* wo     = (const float*)d_in[6];
  const float* gate_w = (const float*)d_in[7];
  const float* w_gate = (const float*)d_in[8];
  const float* w_up   = (const float*)d_in[9];
  const float* w_down = (const float*)d_in[10];
  float* out = (float*)d_out;

  char* p = (char*)d_ws;
  auto alloc = [&](size_t bytes) -> void* {
    void* r = (void*)p;
    p += (bytes + 255) & ~(size_t)255;
    return r;
  };
  uint16_t* xn1  = (uint16_t*)alloc((size_t)T_ * H_ * 2);
  uint16_t* wqt  = (uint16_t*)alloc((size_t)H_ * H_ * 2);
  uint16_t* wkt  = (uint16_t*)alloc((size_t)H_ * (KVH_*HD_) * 2);
  uint16_t* wvt  = (uint16_t*)alloc((size_t)H_ * (KVH_*HD_) * 2);
  uint16_t* wot  = (uint16_t*)alloc((size_t)H_ * H_ * 2);
  float*    qf   = (float*)alloc((size_t)T_ * H_ * 4);
  float*    kf   = (float*)alloc((size_t)T_ * (KVH_*HD_) * 4);
  float*    vf   = (float*)alloc((size_t)T_ * (KVH_*HD_) * 4);
  uint16_t* qbuf = (uint16_t*)alloc((size_t)T_ * H_ * 2);
  uint16_t* kbuf = (uint16_t*)alloc((size_t)T_ * (KVH_*HD_) * 2);
  uint16_t* vbuf = (uint16_t*)alloc((size_t)T_ * (KVH_*HD_) * 2);
  uint16_t* attn = (uint16_t*)alloc((size_t)T_ * H_ * 2);
  float*    x1   = (float*)alloc((size_t)T_ * H_ * 4);
  uint16_t* xn2  = (uint16_t*)alloc((size_t)T_ * H_ * 2);
  int*      cnt      = (int*)alloc(256);
  int*      toklist  = (int*)alloc((size_t)E_ * T_ * 4);
  int*      slotlist = (int*)alloc((size_t)E_ * T_ * 4);
  float*    wlist    = (float*)alloc((size_t)E_ * T_ * 4);
  uint16_t* wgt_e = (uint16_t*)alloc((size_t)H_ * I_ * 2);
  uint16_t* wut_e = (uint16_t*)alloc((size_t)H_ * I_ * 2);
  uint16_t* wdt_e = (uint16_t*)alloc((size_t)H_ * I_ * 2);
  uint16_t* Hg    = (uint16_t*)alloc((size_t)T_ * I_ * 2);
  float*    oslot = (float*)alloc((size_t)2 * T_ * H_ * 4);

  dim3 gblk(32, 8);

  // x -> rmsnorm -> xn1 (bf16)
  rmsnorm_bf16_k<<<T_, 256, 0, stream>>>(x, anw, xn1);
  // weight conversions (transposed bf16)
  convT_bf16_k<<<(H_*H_)/256, 256, 0, stream>>>(wq, wqt, H_, H_);
  convT_bf16_k<<<(H_*KVH_*HD_)/256, 256, 0, stream>>>(wk, wkt, H_, KVH_*HD_);
  convT_bf16_k<<<(H_*KVH_*HD_)/256, 256, 0, stream>>>(wv, wvt, H_, KVH_*HD_);
  convT_bf16_k<<<(H_*H_)/256, 256, 0, stream>>>(wo, wot, H_, H_);
  // QKV projections  (M=2048 -> 64 supertiles of 32 rows -> grid.y = 8)
  gemm_bf16_f32_k<0><<<dim3(H_/64, T_/256), gblk, 0, stream>>>(xn1, wqt, nullptr, qf, T_, H_, H_);
  gemm_bf16_f32_k<0><<<dim3((KVH_*HD_)/64, T_/256), gblk, 0, stream>>>(xn1, wkt, nullptr, kf, T_, KVH_*HD_, H_);
  gemm_bf16_f32_k<0><<<dim3((KVH_*HD_)/64, T_/256), gblk, 0, stream>>>(xn1, wvt, nullptr, vf, T_, KVH_*HD_, H_);
  // RoPE + layouts
  rope_q_k<<<(B_*S_*NH_*32)/256, 256, 0, stream>>>(qf, qbuf);
  rope_k_k<<<(B_*S_*KVH_*32)/256, 256, 0, stream>>>(kf, kbuf);
  vtrans_k<<<(B_*S_*KVH_*HD_)/256, 256, 0, stream>>>(vf, vbuf);
  // attention
  attention_k<<<(B_*NH_*(S_/16))/8, gblk, 0, stream>>>(qbuf, kbuf, vbuf, attn);
  // output proj + residual -> x1
  gemm_bf16_f32_k<1><<<dim3(H_/64, T_/256), gblk, 0, stream>>>(attn, wot, x, x1, T_, H_, H_);
  // second norm
  rmsnorm_bf16_k<<<T_, 256, 0, stream>>>(x1, mnw, xn2);
  // routing
  zero_counts_k<<<1, 32, 0, stream>>>(cnt);
  gate_topk_k<<<T_/8, gblk, 0, stream>>>(xn2, gate_w, cnt, toklist, slotlist, wlist);
  // experts
  for (int e = 0; e < E_; ++e) {
    convT_bf16_k<<<(H_*I_)/256, 256, 0, stream>>>(w_gate + (size_t)e*H_*I_, wgt_e, H_, I_);
    convT_bf16_k<<<(H_*I_)/256, 256, 0, stream>>>(w_up   + (size_t)e*H_*I_, wut_e, H_, I_);
    convT_bf16_k<<<(I_*H_)/256, 256, 0, stream>>>(w_down + (size_t)e*I_*H_, wdt_e, I_, H_);
    gemm_gateup_k<<<dim3(I_/32, T_/256), gblk, 0, stream>>>(xn2, toklist + e*T_, cnt + e,
                                                            wgt_e, wut_e, Hg, I_, H_);
    gemm_down_k<<<dim3(H_/64, T_/256), gblk, 0, stream>>>(Hg, cnt + e, toklist + e*T_,
                                                          slotlist + e*T_, wlist + e*T_,
                                                          wdt_e, oslot, H_, I_);
  }
  // combine
  final_combine_k<<<(T_*H_)/256, 256, 0, stream>>>(x1, oslot, out);
}
